// NaiveGraphFieldConditionalNetwork_10720238371337
// MI455X (gfx1250) — matrix operand order repaired and checked by reading, not compiled, not run on hardware
//
#include <hip/hip_runtime.h>
#include <hip/hip_bf16.h>

typedef _Float16 half_t;
typedef _Float16 v16h __attribute__((ext_vector_type(16)));
typedef _Float16 v8h  __attribute__((ext_vector_type(8)));
typedef _Float16 v2h  __attribute__((ext_vector_type(2)));
typedef float    v8f  __attribute__((ext_vector_type(8)));

#define NATOM 512
#define NJB   16          // number of j-blocks (512/32)
#define EPSF  1e-6f

__device__ __forceinline__ void lds_fence() {
  asm volatile("s_wait_dscnt 0x0" ::: "memory");
}

// ---- CDNA5 async global->LDS copy (ASYNCcnt-tracked, no VGPR round trip) --
__device__ __forceinline__ void async_ld_b128(void* lds_ptr, const void* gptr) {
  // low 32 bits of a generic LDS pointer are the LDS byte offset
  uint32_t lo = (uint32_t)(uintptr_t)lds_ptr;
  asm volatile("global_load_async_to_lds_b128 %0, %1, off"
               :: "v"(lo), "v"(gptr) : "memory");
}
__device__ __forceinline__ void wait_async() {
  asm volatile("s_wait_asynccnt 0x0" ::: "memory");
}

// ---- WMMA operand builders (wave32, f16 16x16x32) -------------------------
// A (16x32): lane&15 = M row, lane>>4 selects K split {0-7,16-23}/{8-15,24-31}
__device__ __forceinline__ v16h load_a_tile(const half_t* st, int kt, int lane) {
  const int row = lane & 15;
  const int hh  = lane >> 4;
  const v8h* p = (const v8h*)(st + row * 64 + kt * 32 + hh * 8);
  v8h a0 = p[0];   // K  +0..7  (of this half's set)
  v8h a1 = p[2];   // K +16..23 (of this half's set)
  return __builtin_shufflevector(a0, a1, 0,1,2,3,4,5,6,7,8,9,10,11,12,13,14,15);
}
// B (32x16): lane&15 = N col, lane>>4 selects K split {0-7,16-23}/{8-15,24-31}; weights transposed [n][k]
__device__ __forceinline__ v16h load_b_tile(const half_t* wt, int kt, int nt, int lane) {
  const int n = nt * 16 + (lane & 15);
  const v8h* p = (const v8h*)(wt + n * 64 + kt * 32 + (lane >> 4) * 8);
  v8h b0 = p[0];
  v8h b1 = p[2];
  return __builtin_shufflevector(b0, b1, 0,1,2,3,4,5,6,7,8,9,10,11,12,13,14,15);
}

// [16 edges,64] @ [64,64] + bias -> 4 C tiles (bias-initialized accumulators)
__device__ __forceinline__ void gemm_16x64(const half_t* Ast, const half_t* Wt,
                                           const float* bias, v8f c[4], int lane) {
  v16h a0 = load_a_tile(Ast, 0, lane);
  v16h a1 = load_a_tile(Ast, 1, lane);
  const int nidx = lane & 15;
#pragma unroll
  for (int nt = 0; nt < 4; ++nt) {
    float bv = bias[nt * 16 + nidx];
    v8f cc = {bv, bv, bv, bv, bv, bv, bv, bv};
    v16h b0 = load_b_tile(Wt, 0, nt, lane);
    cc = __builtin_amdgcn_wmma_f32_16x16x32_f16(false, a0, false, b0, (short)0, cc, false, false);
    v16h b1 = load_b_tile(Wt, 1, nt, lane);
    cc = __builtin_amdgcn_wmma_f32_16x16x32_f16(false, a1, false, b1, (short)0, cc, false, false);
    c[nt] = cc;
  }
}

// 2-layer MLP: relu(X@W1+b1)@W2+b2 for a 16-edge tile, Z restaged via LDS
__device__ __forceinline__ void mlp2_relu(const half_t* Xst, half_t* Zst,
                                          const half_t* W1t, const half_t* W2t,
                                          const float* b1, const float* b2,
                                          v8f out[4], int lane) {
  v8f z[4];
  gemm_16x64(Xst, W1t, b1, z, lane);
  const int nidx = lane & 15, hh = lane >> 4;
#pragma unroll
  for (int nt = 0; nt < 4; ++nt)
#pragma unroll
    for (int r = 0; r < 8; ++r) {
      float v = z[nt][r];
      Zst[(hh * 8 + r) * 64 + nt * 16 + nidx] = (half_t)(v > 0.f ? v : 0.f);
    }
  lds_fence();
  gemm_16x64(Zst, W2t, b2, out, lane);
}

// ---- setup kernels --------------------------------------------------------
__global__ void init_state(const float* __restrict__ pos, const int* __restrict__ atype,
                           const float* __restrict__ atom_emb,
                           float* __restrict__ hi, float* __restrict__ xw) {
  int idx = blockIdx.x * blockDim.x + threadIdx.x;
  if (idx < NATOM * 64) hi[idx] = atom_emb[atype[idx >> 6] * 64 + (idx & 63)];
  if (idx < NATOM * 3)  xw[idx] = pos[idx];
}

// Convert 7 weight mats x 3 layers to f16, transposed: wbf[l][m][n*64+k] = W[k][n]
__global__ void convert_weights(const float* __restrict__ h1, const float* __restrict__ h2,
                                const float* __restrict__ x1, const float* __restrict__ x2,
                                const float* __restrict__ c1, const float* __restrict__ c2,
                                const float* __restrict__ d1, half_t* __restrict__ wout) {
  int idx = blockIdx.x * blockDim.x + threadIdx.x;
  const int TOT = 3 * 7 * 4096;
  if (idx >= TOT) return;
  int l = idx / (7 * 4096);
  int r = idx - l * (7 * 4096);
  int m = r >> 12;
  int e = r & 4095;
  int n = e >> 6, k = e & 63;
  const float* src;
  switch (m) {
    case 0: src = h1; break;
    case 1: src = h2; break;
    case 2: src = x1; break;
    case 3: src = x2; break;
    case 4: src = c1; break;
    case 5: src = c2; break;
    default: src = d1; break;
  }
  wout[idx] = (half_t)src[l * 4096 + k * 64 + n];
}

// ---- per-layer node prep: hn = hi/||hi+eps(1-am)|| + eps ; y0 = hn @ Win ---
__global__ void node_prep(const float* __restrict__ hi, const int* __restrict__ atype,
                          const float* __restrict__ Win, int layer,
                          float* __restrict__ hn, float* __restrict__ y0) {
  const int i = blockIdx.x;
  const int t = threadIdx.x;   // 64 threads (2 waves)
  __shared__ float srow[64];
  __shared__ float spart[2];
  float v  = hi[i * 64 + t];
  float am = (atype[i] > 0) ? 1.f : 0.f;
  float adj = (1.f - am) * EPSF;
  float s = (v + adj) * (v + adj);
#pragma unroll
  for (int m = 16; m >= 1; m >>= 1) s += __shfl_xor(s, m, 32);
  if ((t & 31) == 0) spart[t >> 5] = s;
  __syncthreads();
  float nrm = sqrtf(spart[0] + spart[1]);
  float hv = v / nrm + EPSF;
  hn[i * 64 + t] = hv;
  srow[t] = hv;
  __syncthreads();
  const float* W = Win + layer * 4096;
  float acc = 0.f;
  for (int k = 0; k < 64; ++k) acc += srow[k] * W[k * 64 + t];
  y0[i * 64 + t] = acc;
}

// ---- fused edge kernel: all per-edge MLPs via WMMA ------------------------
// Grid: (NATOM/16, NATOM/32). Block 128 threads (4 waves).
// Wave w handles i rows [i0+4w, i0+4w+4) x 32 j's -> 8 M-tiles of 16 edges.
__launch_bounds__(128)
__global__ void edge_kernel(
    const float* __restrict__ hn, const float* __restrict__ y0,
    const float* __restrict__ xw, const int* __restrict__ atype,
    const int* __restrict__ btype, const float* __restrict__ bond_emb,
    const half_t* __restrict__ wbf,
    const float* __restrict__ b_h1, const float* __restrict__ b_h2,
    const float* __restrict__ b_x1, const float* __restrict__ b_x2,
    const float* __restrict__ b_c1, const float* __restrict__ b_c2,
    const float* __restrict__ b_d1, const float* __restrict__ w_d2,
    const float* __restrict__ b_d2,
    float* __restrict__ pY, float* __restrict__ pX, int layer) {
  __shared__ __align__(16) half_t sW[7 * 4096];    // 7 transposed f16 weight mats
  __shared__ __align__(16) float sY[32 * 64];      // y0 for this j-block
  __shared__ float sXj[32 * 3];
  __shared__ float sAmJ[32];
  __shared__ __align__(16) half_t sBond[4 * 64];
  __shared__ __align__(16) half_t sStX[4][1024];   // per-wave input staging
  __shared__ __align__(16) half_t sStZ[4][1024];   // per-wave hidden staging
  __shared__ float sR[4][16];
  __shared__ float sEm[4][16];
  __shared__ float sPhi[4][16];
  __shared__ int   sBt[4][16];

  const int tid  = threadIdx.x;
  const int wave = tid >> 5;
  const int lane = tid & 31;
  const int i0 = blockIdx.x * 16;
  const int j0 = blockIdx.y * 32;
  const int jblk = blockIdx.y;

  // -- async bulk copies: weights (57344B) + y0 block (8192B) --
  {
    const uint4* wsrc4 = (const uint4*)(wbf + layer * 7 * 4096);
    uint4* wdst4 = (uint4*)sW;
    for (int idx = tid; idx < 7 * 4096 / 8; idx += 128)   // 3584 x 16B
      async_ld_b128(wdst4 + idx, wsrc4 + idx);
    const uint4* ysrc4 = (const uint4*)(y0 + j0 * 64);
    uint4* ydst4 = (uint4*)sY;
    for (int idx = tid; idx < 32 * 64 / 4; idx += 128)    // 512 x 16B
      async_ld_b128(ydst4 + idx, ysrc4 + idx);
  }
  // -- small direct copies --
  for (int idx = tid; idx < 96; idx += 128) sXj[idx] = xw[j0 * 3 + idx];
  if (tid < 32) sAmJ[tid] = (atype[j0 + tid] > 0) ? 1.f : 0.f;
  for (int idx = tid; idx < 256; idx += 128) sBond[idx] = (half_t)bond_emb[layer * 256 + idx];
  wait_async();
  __syncthreads();

  const int nidx  = lane & 15;
  const int halfi = lane >> 4;
  half_t* stX = sStX[wave];
  half_t* stZ = sStZ[wave];
  const float db2 = b_d2[layer];

  for (int ii = 0; ii < 4; ++ii) {
    const int i = i0 + wave * 4 + ii;
    const float am_i = (atype[i] > 0) ? 1.f : 0.f;
    const float xi0 = xw[i * 3 + 0], xi1 = xw[i * 3 + 1], xi2 = xw[i * 3 + 2];
    float accY[4] = {0.f, 0.f, 0.f, 0.f};
    float accX0 = 0.f, accX1 = 0.f, accX2 = 0.f;

    for (int jt = 0; jt < 2; ++jt) {
      const int jb = jt * 16;
      // per-row geometry + bond type (lanes 0..15 each own one edge row)
      if (lane < 16) {
        int jl = jb + lane;
        int j  = j0 + jl;
        float em  = (am_i > 0.f && sAmJ[jl] > 0.f && i != j) ? 1.f : 0.f;
        float adj = (1.f - em) * EPSF;
        float d0 = sXj[jl * 3 + 0] - xi0 + adj;
        float d1 = sXj[jl * 3 + 1] - xi1 + adj;
        float d2 = sXj[jl * 3 + 2] - xi2 + adj;
        sR[wave][lane]  = sqrtf(d0 * d0 + d1 * d1 + d2 * d2);
        sEm[wave][lane] = em;
        sBt[wave][lane] = btype[i * NATOM + j];
      }
      lds_fence();

      v8f prod[4], tmp[4];
      // ---- encoder h: 0.5*(hn[j]+hn[i]) (packed v2h stores) ----
      for (int idx = lane; idx < 512; idx += 32) {
        int r = idx >> 5, kp = (idx & 31) * 2;
        int j = j0 + jb + r;
        float2 a = *(const float2*)(hn + j * 64 + kp);
        float2 b = *(const float2*)(hn + i * 64 + kp);
        v2h o = { (half_t)(0.5f * (a.x + b.x)), (half_t)(0.5f * (a.y + b.y)) };
        *(v2h*)(stX + r * 64 + kp) = o;
      }
      lds_fence();
      mlp2_relu(stX, stZ, sW + 0 * 4096, sW + 1 * 4096,
                b_h1 + layer * 64, b_h2 + layer * 64, prod, lane);

      // ---- encoder x: Gaussian RBF of r_ij ----
      for (int idx = lane; idx < 512; idx += 32) {
        int r = idx >> 5, kp = (idx & 31) * 2;
        float rr = sR[wave][r];
        float t0 = (rr - (float)kp * (10.0f / 63.0f)) * (1.0f / 0.3f);
        float t1 = (rr - (float)(kp + 1) * (10.0f / 63.0f)) * (1.0f / 0.3f);
        v2h o = { (half_t)__expf(-0.5f * t0 * t0), (half_t)__expf(-0.5f * t1 * t1) };
        *(v2h*)(stX + r * 64 + kp) = o;
      }
      lds_fence();
      mlp2_relu(stX, stZ, sW + 2 * 4096, sW + 3 * 4096,
                b_x1 + layer * 64, b_x2 + layer * 64, tmp, lane);
#pragma unroll
      for (int nt = 0; nt < 4; ++nt) prod[nt] *= tmp[nt];

      // ---- encoder c: bond embedding ----
      for (int idx = lane; idx < 512; idx += 32) {
        int r = idx >> 5, kp = (idx & 31) * 2;
        int bt = sBt[wave][r];
        *(v2h*)(stX + r * 64 + kp) = *(const v2h*)(sBond + bt * 64 + kp);
      }
      lds_fence();
      mlp2_relu(stX, stZ, sW + 4 * 4096, sW + 5 * 4096,
                b_c1 + layer * 64, b_c2 + layer * 64, tmp, lane);
#pragma unroll
      for (int nt = 0; nt < 4; ++nt) prod[nt] *= tmp[nt];

      // ---- mask, cfconv partial, restage m for decoder ----
#pragma unroll
      for (int nt = 0; nt < 4; ++nt) {
#pragma unroll
        for (int r = 0; r < 8; ++r) {
          int row = halfi * 8 + r;
          float m = prod[nt][r] * sEm[wave][row];
          accY[nt] += m * sY[(jb + row) * 64 + nt * 16 + nidx];
          stX[row * 64 + nt * 16 + nidx] = (half_t)m;
        }
      }
      lds_fence();

      // ---- decoder: phi = relu(m@dW1+db1)@dW2 + db2 ----
      v8f z2[4];
      gemm_16x64(stX, sW + 6 * 4096, b_d1 + layer * 64, z2, lane);
      float w2v[4];
#pragma unroll
      for (int nt = 0; nt < 4; ++nt) w2v[nt] = w_d2[layer * 64 + nt * 16 + nidx];
      float p[8];
#pragma unroll
      for (int r = 0; r < 8; ++r) {
        float s = 0.f;
#pragma unroll
        for (int nt = 0; nt < 4; ++nt) {
          float zv = z2[nt][r];
          s += (zv > 0.f ? zv : 0.f) * w2v[nt];
        }
        p[r] = s;
      }
#pragma unroll
      for (int m = 1; m <= 8; m <<= 1)
#pragma unroll
        for (int r = 0; r < 8; ++r) p[r] += __shfl_xor(p[r], m, 32);
      if (nidx == 0) {
#pragma unroll
        for (int r = 0; r < 8; ++r) sPhi[wave][halfi * 8 + r] = p[r] + db2;
      }
      lds_fence();

      // ---- coordinate contribution: em/r * d_ij * phi ----
      float cx0 = 0.f, cx1 = 0.f, cx2 = 0.f;
      if (lane < 16) {
        float em = sEm[wave][lane];
        if (em > 0.f) {
          int jl = jb + lane;
          float coef = sPhi[wave][lane] / sR[wave][lane];
          cx0 = coef * (sXj[jl * 3 + 0] - xi0);
          cx1 = coef * (sXj[jl * 3 + 1] - xi1);
          cx2 = coef * (sXj[jl * 3 + 2] - xi2);
        }
      }
      accX0 += cx0; accX1 += cx1; accX2 += cx2;
    } // jt

    // ---- write exclusive per-(j-block, i) partials (deterministic) ----
#pragma unroll
    for (int nt = 0; nt < 4; ++nt) accY[nt] += __shfl_xor(accY[nt], 16, 32);
    if (lane < 16) {
#pragma unroll
      for (int nt = 0; nt < 4; ++nt)
        pY[(jblk * NATOM + i) * 64 + nt * 16 + nidx] = accY[nt];
    }
#pragma unroll
    for (int m = 1; m <= 8; m <<= 1) {
      accX0 += __shfl_xor(accX0, m, 32);
      accX1 += __shfl_xor(accX1, m, 32);
      accX2 += __shfl_xor(accX2, m, 32);
    }
    if (lane == 0) {
      pX[(jblk * NATOM + i) * 3 + 0] = accX0;
      pX[(jblk * NATOM + i) * 3 + 1] = accX1;
      pX[(jblk * NATOM + i) * 3 + 2] = accX2;
    }
  } // ii
}

// ---- node update: hi += ssp-MLP(sum_j m*y) * am ; xi += sum_j ... ---------
__global__ void node_update(float* __restrict__ hi, float* __restrict__ xw,
                            const float* __restrict__ pY, const float* __restrict__ pX,
                            const float* __restrict__ W1, const float* __restrict__ b1,
                            const float* __restrict__ W2, const float* __restrict__ b2,
                            const int* __restrict__ atype, int layer, int last,
                            const float* __restrict__ pos, float* __restrict__ out) {
  const int i = blockIdx.x;
  const int n = threadIdx.x;  // 64 threads
  __shared__ float y[64];
  __shared__ float u[64];
  float acc = 0.f;
  for (int b = 0; b < NJB; ++b) acc += pY[(b * NATOM + i) * 64 + n];
  y[n] = acc;
  __syncthreads();
  const float* w1 = W1 + layer * 4096;
  float a = b1[layer * 64 + n];
  for (int k = 0; k < 64; ++k) a += y[k] * w1[k * 64 + n];
  a = __logf(1.f + __expf(a)) - 0.6931471805599453f;  // shifted softplus
  u[n] = a;
  __syncthreads();
  const float* w2 = W2 + layer * 4096;
  float v = b2[layer * 64 + n];
  for (int k = 0; k < 64; ++k) v += u[k] * w2[k * 64 + n];
  float am = (atype[i] > 0) ? 1.f : 0.f;
  hi[i * 64 + n] += v * am;
  if (n < 3) {
    float ax = 0.f;
    for (int b = 0; b < NJB; ++b) ax += pX[(b * NATOM + i) * 3 + n];
    float x = xw[i * 3 + n] + ax;
    xw[i * 3 + n] = x;
    if (last) out[i * 3 + n] = x - pos[i * 3 + n];
  }
}

extern "C" void kernel_launch(void* const* d_in, const int* in_sizes, int n_in,
                              void* d_out, int out_size, void* d_ws, size_t ws_size,
                              hipStream_t stream) {
  const float* pos      = (const float*)d_in[0];
  const int*   atype    = (const int*)d_in[1];
  const int*   btype    = (const int*)d_in[2];
  const float* atom_emb = (const float*)d_in[3];
  const float* bond_emb = (const float*)d_in[4];
  const float* ench_W1  = (const float*)d_in[5];
  const float* ench_b1  = (const float*)d_in[6];
  const float* ench_W2  = (const float*)d_in[7];
  const float* ench_b2  = (const float*)d_in[8];
  const float* encx_W1  = (const float*)d_in[9];
  const float* encx_b1  = (const float*)d_in[10];
  const float* encx_W2  = (const float*)d_in[11];
  const float* encx_b2  = (const float*)d_in[12];
  const float* encc_W1  = (const float*)d_in[13];
  const float* encc_b1  = (const float*)d_in[14];
  const float* encc_W2  = (const float*)d_in[15];
  const float* encc_b2  = (const float*)d_in[16];
  const float* sch_Win  = (const float*)d_in[17];
  const float* sch_W1   = (const float*)d_in[18];
  const float* sch_b1   = (const float*)d_in[19];
  const float* sch_W2   = (const float*)d_in[20];
  const float* sch_b2   = (const float*)d_in[21];
  const float* dec_W1   = (const float*)d_in[22];
  const float* dec_b1   = (const float*)d_in[23];
  const float* dec_W2   = (const float*)d_in[24];
  const float* dec_b2   = (const float*)d_in[25];
  (void)in_sizes; (void)n_in; (void)out_size; (void)ws_size;

  float* ws   = (float*)d_ws;
  float* hi   = ws;                    // 512*64
  float* hn   = hi + NATOM * 64;       // 512*64
  float* y0   = hn + NATOM * 64;       // 512*64
  float* xw   = y0 + NATOM * 64;       // 512*3
  float* pY   = xw + NATOM * 3;        // 16*512*64 partials
  float* pX   = pY + NJB * NATOM * 64; // 16*512*3 partials
  half_t* wbf = (half_t*)(pX + NJB * NATOM * 3);  // 3*7*4096 f16 (transposed)
  float* out  = (float*)d_out;

  init_state<<<(NATOM * 64 + 255) / 256, 256, 0, stream>>>(pos, atype, atom_emb, hi, xw);
  convert_weights<<<(3 * 7 * 4096 + 255) / 256, 256, 0, stream>>>(
      ench_W1, ench_W2, encx_W1, encx_W2, encc_W1, encc_W2, dec_W1, wbf);

  dim3 gedge(NATOM / 16, NATOM / 32);
  for (int l = 0; l < 3; ++l) {
    node_prep<<<NATOM, 64, 0, stream>>>(hi, atype, sch_Win, l, hn, y0);
    edge_kernel<<<gedge, 128, 0, stream>>>(
        hn, y0, xw, atype, btype, bond_emb, wbf,
        ench_b1, ench_b2, encx_b1, encx_b2, encc_b1, encc_b2,
        dec_b1, dec_W2, dec_b2, pY, pX, l);
    node_update<<<NATOM, 64, 0, stream>>>(hi, xw, pY, pX,
        sch_W1, sch_b1, sch_W2, sch_b2, atype, l, (l == 2) ? 1 : 0, pos, out);
  }
}
